// BranchingAttention_90787018703326
// MI455X (gfx1250) — matrix operand order repaired and checked by reading, not compiled, not run on hardware
//
#include <hip/hip_runtime.h>

typedef __attribute__((ext_vector_type(16))) __bf16        v16bf;
typedef __attribute__((ext_vector_type(8)))  float         f32x8;
typedef __attribute__((ext_vector_type(4)))  unsigned int  u32x4;
typedef __attribute__((ext_vector_type(8)))  int           i32x8;
typedef __attribute__((ext_vector_type(4)))  int           i32x4;

#define DEVFN static __device__ __forceinline__

#if __has_builtin(__builtin_amdgcn_tensor_load_to_lds) && __has_builtin(__builtin_amdgcn_s_wait_tensorcnt)
#define HAVE_TDM 1
#else
#define HAVE_TDM 0
#endif

constexpr int   B_   = 2, NB_ = 20, BS_ = 64, D_ = 768, H_ = 12, DH_ = 64;
constexpr int   D3_  = 3 * D_;                 // 2304
constexpr int   TOK_ = B_ * NB_ * BS_;         // 2560 rows per branch
constexpr int   M_   = 2 * TOK_;               // 5120 rows (both branches)
constexpr float NEGV  = -1.0e9f;
constexpr float SCALE = 0.125f;                // 1/sqrt(DH)

// workspace layout (units: bf16 halves). All offsets are multiples of 8 -> 16B aligned.
constexpr size_t OFF_XB  = 0;                                  // x (both branches) bf16 [M_][D_]
constexpr size_t OFF_WA  = OFF_XB  + (size_t)M_  * D_;         // w_attn^T bf16 [D3_][D_]
constexpr size_t OFF_WP  = OFF_WA  + (size_t)D3_ * D_;         // w_proj^T bf16 [D_][D_]
constexpr size_t OFF_QKV = OFF_WP  + (size_t)D_  * D_;         // qkv bf16 [M_][D3_] (v|q|k)
constexpr size_t OFF_AO  = OFF_QKV + (size_t)M_  * D3_;        // attn out bf16 [M_][D_]

DEVFN unsigned short f2bf(float f) {            // round-to-nearest-even fp32 -> bf16
  unsigned u = __builtin_bit_cast(unsigned, f);
  u += 0x7FFFu + ((u >> 16) & 1u);
  return (unsigned short)(u >> 16);
}

union FragBF { u32x4 u[2]; v16bf v; };          // 32B: one WMMA A/B operand per lane

DEVFN f32x8 zero8() { f32x8 z = {0.f,0.f,0.f,0.f,0.f,0.f,0.f,0.f}; return z; }

#if HAVE_TDM
// ---------------------------------------------------------------------------
// TDM: async 2D bf16 tile load Global -> LDS with hardware row padding.
// pad_ivl: 3 => pad every 16 DW (32 bf16 row), 4 => every 32 DW (64 bf16 row)
// pad_amt: 3 => insert 4 DW (8 halves) of padding
// ---------------------------------------------------------------------------
DEVFN void tdm_load_tile_bf16(void* lds_dst, const void* gsrc,
                              int tile_w, int tile_h, int row_stride,
                              int pad_ivl, int pad_amt)
{
  const unsigned long long ga = (unsigned long long)(uintptr_t)gsrc;
  u32x4 g0;
  g0[0] = 1u;                                                  // count=1 (valid user D#)
  g0[1] = (unsigned)(uintptr_t)lds_dst;                        // lds_addr (low 32 = LDS offset)
  g0[2] = (unsigned)ga;                                        // global_addr[31:0]
  g0[3] = (unsigned)((ga >> 32) & 0x1FFFFFFull) | (2u << 30);  // global_addr[56:32] | type=2
  i32x8 g1;
  g1[0] = (1 << 16)                    // data_size = 2 bytes (bf16)
        | (1 << 20)                    // pad_enable
        | (pad_ivl << 22) | (pad_amt << 25);
  g1[1] = (tile_w & 0xFFFF) << 16;     // tensor_dim0[15:0]  (tile always in-bounds)
  g1[2] = (tile_h & 0xFFFF) << 16;     // tensor_dim1[15:0]
  g1[3] = (tile_w & 0xFFFF) << 16;     // tile_dim0
  g1[4] = (tile_h & 0xFFFF);           // tile_dim1 (tile_dim2 = 0)
  g1[5] = row_stride;                  // tensor_dim0_stride[31:0] (elements)
  g1[6] = 0;
  g1[7] = 0;
  const i32x4 z4 = {0, 0, 0, 0};
#if __clang_major__ >= 23
  const i32x8 z8 = {0, 0, 0, 0, 0, 0, 0, 0};
  __builtin_amdgcn_tensor_load_to_lds(g0, g1, z4, z4, z8, 0);
#else
  __builtin_amdgcn_tensor_load_to_lds(g0, g1, z4, z4, 0);
#endif
}
#endif

// ---------------------------------------------------------------------------
// Kernel 1: fp32 -> bf16 conversion; weights transposed to [N][K]
// ---------------------------------------------------------------------------
__global__ __launch_bounds__(256) void convert_kernel(
    const float* __restrict__ x0, const float* __restrict__ x1,
    const float* __restrict__ wa, const float* __restrict__ wp,
    unsigned short* __restrict__ ws)
{
  const size_t stride = (size_t)gridDim.x * blockDim.x;
  const size_t t0 = (size_t)blockIdx.x * blockDim.x + threadIdx.x;
  for (size_t i = t0; i < (size_t)TOK_ * D_; i += stride) {
    ws[OFF_XB + i]                     = f2bf(x0[i]);
    ws[OFF_XB + (size_t)TOK_ * D_ + i] = f2bf(x1[i]);
  }
  for (size_t i = t0; i < (size_t)D3_ * D_; i += stride) {
    const size_t n = i / D_, k = i % D_;
    ws[OFF_WA + i] = f2bf(wa[k * D3_ + n]);
  }
  for (size_t i = t0; i < (size_t)D_ * D_; i += stride) {
    const size_t n = i / D_, k = i % D_;
    ws[OFF_WP + i] = f2bf(wp[k * D_ + n]);
  }
}

// ---------------------------------------------------------------------------
// Kernel 2/4: bf16 GEMM  C = A * Bt^T + bias.  A:[M][K], Bt:[N][K] (pre-transposed).
// 128x128 tile, BK=32, 8 waves, each wave a 32x64 region (2x4 WMMA tiles).
// TDM double-buffered staging: next k-block streams while current one computes.
// ---------------------------------------------------------------------------
template<int OUTF32>
__global__ __launch_bounds__(256) void gemm_kernel(
    const unsigned short* __restrict__ A,
    const unsigned short* __restrict__ Bt,
    const float* __restrict__ bias,
    unsigned short* __restrict__ Cbf,
    float* __restrict__ Cf,
    int M, int N, int K)
{
  __shared__ __align__(16) unsigned short As[2][128][40];  // 32 + 8 pad halves
  __shared__ __align__(16) unsigned short Bs[2][128][40];

  const int nbase = blockIdx.x * 128;
  const int mbase = blockIdx.y * 128;
  const int tid   = threadIdx.x;
  const int lane  = tid & 31, wave = tid >> 5;
  const int lr    = lane & 15, hi = lane >> 4;
  const int wm    = (wave & 3) * 32;     // wave's M offset in tile
  const int wn    = (wave >> 2) * 64;    // wave's N offset in tile

  f32x8 acc[2][4];
  #pragma unroll
  for (int mt = 0; mt < 2; ++mt)
    #pragma unroll
    for (int nt = 0; nt < 4; ++nt) acc[mt][nt] = zero8();

#if HAVE_TDM
  if (wave == 0) {   // prologue: stream first k-block
    tdm_load_tile_bf16(&As[0][0][0], A  + (size_t)mbase * K, 32, 128, K, 3, 3);
    tdm_load_tile_bf16(&Bs[0][0][0], Bt + (size_t)nbase * K, 32, 128, K, 3, 3);
  }
#endif

  int buf = 0;
  for (int k0 = 0; k0 < K; k0 += 32, buf ^= 1) {
#if HAVE_TDM
    if (wave == 0) __builtin_amdgcn_s_wait_tensorcnt(0);   // tile(buf) landed in LDS
    __syncthreads();
    if (wave == 0 && k0 + 32 < K) {                        // prefetch tile(buf^1)
      tdm_load_tile_bf16(&As[buf ^ 1][0][0], A  + (size_t)mbase * K + k0 + 32, 32, 128, K, 3, 3);
      tdm_load_tile_bf16(&Bs[buf ^ 1][0][0], Bt + (size_t)nbase * K + k0 + 32, 32, 128, K, 3, 3);
    }
#else
    {
      const int ldrow = tid >> 1;
      const int ldcol = (tid & 1) * 16;
      const u32x4* ag = (const u32x4*)(A  + (size_t)(mbase + ldrow) * K + k0 + ldcol);
      const u32x4* bg = (const u32x4*)(Bt + (size_t)(nbase + ldrow) * K + k0 + ldcol);
      const u32x4 a0 = ag[0], a1 = ag[1];
      const u32x4 b0 = bg[0], b1 = bg[1];
      __syncthreads();
      *(u32x4*)&As[buf][ldrow][ldcol]     = a0;
      *(u32x4*)&As[buf][ldrow][ldcol + 8] = a1;
      *(u32x4*)&Bs[buf][ldrow][ldcol]     = b0;
      *(u32x4*)&Bs[buf][ldrow][ldcol + 8] = b1;
      __syncthreads();
    }
#endif

    FragBF af[2], bf[4];
    #pragma unroll
    for (int mt = 0; mt < 2; ++mt) {                  // A frag: row=lr, K=16*hi+0..15
      const int r = wm + mt * 16 + lr;
      af[mt].u[0] = *(const u32x4*)&As[buf][r][16 * hi];
      af[mt].u[1] = *(const u32x4*)&As[buf][r][16 * hi + 8];
    }
    #pragma unroll
    for (int nt = 0; nt < 4; ++nt) {                  // B frag: col=lr, K=16*hi+0..15
      const int r = wn + nt * 16 + lr;
      bf[nt].u[0] = *(const u32x4*)&Bs[buf][r][16 * hi];
      bf[nt].u[1] = *(const u32x4*)&Bs[buf][r][16 * hi + 8];
    }
    #pragma unroll
    for (int mt = 0; mt < 2; ++mt)
      #pragma unroll
      for (int nt = 0; nt < 4; ++nt)
        acc[mt][nt] = __builtin_amdgcn_wmma_f32_16x16x32_bf16(
            false, af[mt].v, false, bf[nt].v, (short)0, acc[mt][nt], false, false);

    __syncthreads();   // all waves done reading buf before it is overwritten
  }

  #pragma unroll
  for (int mt = 0; mt < 2; ++mt)
    #pragma unroll
    for (int nt = 0; nt < 4; ++nt)
      #pragma unroll
      for (int r = 0; r < 8; ++r) {
        const int m = mbase + wm + mt * 16 + r + 8 * hi;   // C/D layout: M=r+8*hi, N=lane&15
        const int n = nbase + wn + nt * 16 + lr;
        const float v = acc[mt][nt][r] + bias[n];
        if (OUTF32) Cf[(size_t)m * N + n]  = v;
        else        Cbf[(size_t)m * N + n] = f2bf(v);
      }
}

// ---------------------------------------------------------------------------
// Kernel 3: flash-style branching attention.
// grid = (NB, H, 2*B): one WG (4 waves) per (branch, batch, head, query block i).
// Streams K0/V0 blocks j<i (branch 0) then the causal self block.
// K tile staged by TDM; V tile transposed manually (TDM has no transpose) while
// the DMA streams, so the two overlap.
// ---------------------------------------------------------------------------
__global__ __launch_bounds__(128) void attn_kernel(
    const unsigned short* __restrict__ qkv,   // [M_][D3_] bf16 : v|q|k
    unsigned short* __restrict__ aout)        // [M_][D_]  bf16 (merged heads)
{
  const int i  = blockIdx.x;
  const int h  = blockIdx.y;
  const int br = blockIdx.z >> 1;
  const int b  = blockIdx.z & 1;
  const int hh = h * DH_;

  const int tid  = threadIdx.x;
  const int lane = tid & 31, wave = tid >> 5;
  const int lr   = lane & 15, hi = lane >> 4;

  __shared__ __align__(16) unsigned short Ks[64][72];      // [key][dh], padded (TDM pads)
  __shared__ __align__(16) unsigned short Vt[64][72];      // [dh][key] (transposed V)
  __shared__ __align__(16) unsigned short Ps[4][16][72];   // per-wave P spill (16x64)

  const size_t selfbase = ((size_t)((br * B_ + b) * NB_ + i)) * BS_;
  const size_t b0base   = ((size_t)(b * NB_)) * BS_;       // branch 0, batch b

  // Q fragments (A layout), loaded straight from global. Wave owns rows 16w..16w+15.
  FragBF aQ[2];
  {
    const size_t qoff = (selfbase + wave * 16 + lr) * (size_t)D3_ + D_ + hh;
    #pragma unroll
    for (int s = 0; s < 2; ++s) {
      aQ[s].u[0] = *(const u32x4*)(qkv + qoff + 32 * s + 16 * hi);
      aQ[s].u[1] = *(const u32x4*)(qkv + qoff + 32 * s + 16 * hi + 8);
    }
  }

  f32x8 accO[4];
  #pragma unroll
  for (int nt = 0; nt < 4; ++nt) accO[nt] = zero8();
  float m_i[8], l_i[8];
  #pragma unroll
  for (int r = 0; r < 8; ++r) { m_i[r] = -3.0e38f; l_i[r] = 0.f; }

  for (int j = 0; j <= i; ++j) {
    const bool self = (j == i);
    const size_t rb = self ? selfbase : b0base + (size_t)j * BS_;

    { // stage K (TDM) and V (manual transpose) into LDS
      const int row = tid >> 1;
      const int seg = (tid & 1) * 32;
      const unsigned short* vp = qkv + (rb + row) * (size_t)D3_ + hh + seg;
      u32x4 vq[4];
      #pragma unroll
      for (int q = 0; q < 4; ++q) vq[q] = ((const u32x4*)vp)[q];
#if !HAVE_TDM
      const unsigned short* kp = qkv + (rb + row) * (size_t)D3_ + 2 * D_ + hh + seg;
      u32x4 kq[4];
      #pragma unroll
      for (int q = 0; q < 4; ++q) kq[q] = ((const u32x4*)kp)[q];
#endif
      __syncthreads();                       // previous iteration done reading LDS
#if HAVE_TDM
      if (wave == 0)                         // 64x64 bf16 tile, rows padded 64->72 halves
        tdm_load_tile_bf16(&Ks[0][0], qkv + rb * (size_t)D3_ + 2 * D_ + hh,
                           64, 64, D3_, 4, 3);
#else
      #pragma unroll
      for (int q = 0; q < 4; ++q) *(u32x4*)&Ks[row][seg + 8 * q] = kq[q];
#endif
      #pragma unroll
      for (int q = 0; q < 4; ++q) {
        union { u32x4 u; unsigned short s[8]; } t; t.u = vq[q];
        #pragma unroll
        for (int e = 0; e < 8; ++e) Vt[seg + 8 * q + e][row] = t.s[e];
      }
#if HAVE_TDM
      if (wave == 0) __builtin_amdgcn_s_wait_tensorcnt(0);
#endif
      __syncthreads();
    }

    // S = Q * K^T  (16 queries x 64 keys per wave), 2 K-steps of 32 over dh
    f32x8 sc[4];
    #pragma unroll
    for (int nt = 0; nt < 4; ++nt) {
      FragBF k0f, k1f;
      const int r = nt * 16 + lr;
      k0f.u[0] = *(const u32x4*)&Ks[r][16 * hi];
      k0f.u[1] = *(const u32x4*)&Ks[r][16 * hi + 8];
      k1f.u[0] = *(const u32x4*)&Ks[r][32 + 16 * hi];
      k1f.u[1] = *(const u32x4*)&Ks[r][32 + 16 * hi + 8];
      f32x8 c = zero8();
      c = __builtin_amdgcn_wmma_f32_16x16x32_bf16(false, aQ[0].v, false, k0f.v, (short)0, c, false, false);
      c = __builtin_amdgcn_wmma_f32_16x16x32_bf16(false, aQ[1].v, false, k1f.v, (short)0, c, false, false);
      sc[nt] = c;
    }

    // scale + causal mask on the self block
    #pragma unroll
    for (int nt = 0; nt < 4; ++nt)
      #pragma unroll
      for (int r = 0; r < 8; ++r) {
        float v = sc[nt][r] * SCALE;
        if (self) {
          const int mrow = wave * 16 + r + 8 * hi;
          const int ncol = nt * 16 + lr;
          if (ncol > mrow) v = NEGV;
        }
        sc[nt][r] = v;
      }

    // online softmax: xor-shuffles 1/2/4/8 reduce within each 16-lane half
    #pragma unroll
    for (int r = 0; r < 8; ++r) {
      float x = fmaxf(fmaxf(sc[0][r], sc[1][r]), fmaxf(sc[2][r], sc[3][r]));
      x = fmaxf(x, __shfl_xor(x, 1, 32));
      x = fmaxf(x, __shfl_xor(x, 2, 32));
      x = fmaxf(x, __shfl_xor(x, 4, 32));
      x = fmaxf(x, __shfl_xor(x, 8, 32));
      const float mnew = fmaxf(m_i[r], x);
      const float corr = __expf(m_i[r] - mnew);
      float rs = 0.f;
      #pragma unroll
      for (int nt = 0; nt < 4; ++nt) {
        const float p = __expf(sc[nt][r] - mnew);
        sc[nt][r] = p;
        rs += p;
      }
      rs += __shfl_xor(rs, 1, 32);
      rs += __shfl_xor(rs, 2, 32);
      rs += __shfl_xor(rs, 4, 32);
      rs += __shfl_xor(rs, 8, 32);
      l_i[r] = l_i[r] * corr + rs;
      m_i[r] = mnew;
      #pragma unroll
      for (int nt = 0; nt < 4; ++nt) accO[nt][r] *= corr;
    }

    // spill P (D layout) to per-wave LDS, reload as A fragments (same-wave DS is in-order)
    #pragma unroll
    for (int nt = 0; nt < 4; ++nt)
      #pragma unroll
      for (int r = 0; r < 8; ++r)
        Ps[wave][r + 8 * hi][nt * 16 + lr] = f2bf(sc[nt][r]);

#if __has_builtin(__builtin_amdgcn_s_wait_dscnt)
    __builtin_amdgcn_s_wait_dscnt(0);
#endif

    FragBF aP[2];
    #pragma unroll
    for (int s = 0; s < 2; ++s) {
      aP[s].u[0] = *(const u32x4*)&Ps[wave][lr][32 * s + 16 * hi];
      aP[s].u[1] = *(const u32x4*)&Ps[wave][lr][32 * s + 16 * hi + 8];
    }

    // O += P * V   (B frags contiguous thanks to transposed V in LDS)
    #pragma unroll
    for (int nt = 0; nt < 4; ++nt) {
      FragBF v0f, v1f;
      const int r = nt * 16 + lr;
      v0f.u[0] = *(const u32x4*)&Vt[r][16 * hi];
      v0f.u[1] = *(const u32x4*)&Vt[r][16 * hi + 8];
      v1f.u[0] = *(const u32x4*)&Vt[r][32 + 16 * hi];
      v1f.u[1] = *(const u32x4*)&Vt[r][32 + 16 * hi + 8];
      accO[nt] = __builtin_amdgcn_wmma_f32_16x16x32_bf16(false, aP[0].v, false, v0f.v, (short)0, accO[nt], false, false);
      accO[nt] = __builtin_amdgcn_wmma_f32_16x16x32_bf16(false, aP[1].v, false, v1f.v, (short)0, accO[nt], false, false);
    }
  }

  // finalize: O /= l, store merged-head bf16
  #pragma unroll
  for (int r = 0; r < 8; ++r) {
    const float inv = 1.0f / l_i[r];
    #pragma unroll
    for (int nt = 0; nt < 4; ++nt) {
      const int row = wave * 16 + r + 8 * hi;
      const int n   = nt * 16 + lr;
      aout[(selfbase + row) * (size_t)D_ + hh + n] = f2bf(accO[nt][r] * inv);
    }
  }
}

// ---------------------------------------------------------------------------
extern "C" void kernel_launch(void* const* d_in, const int* in_sizes, int n_in,
                              void* d_out, int out_size, void* d_ws, size_t ws_size,
                              hipStream_t stream) {
  (void)in_sizes; (void)n_in; (void)out_size; (void)ws_size;
  const float* x0  = (const float*)d_in[0];
  const float* x1  = (const float*)d_in[1];
  const float* wa  = (const float*)d_in[2];
  const float* ba  = (const float*)d_in[3];
  const float* wpj = (const float*)d_in[4];
  const float* bp  = (const float*)d_in[5];
  unsigned short* ws = (unsigned short*)d_ws;
  float* out = (float*)d_out;

  convert_kernel<<<1024, 256, 0, stream>>>(x0, x1, wa, wpj, ws);

  // QKV: [5120 x 768] @ [768 x 2304] + b_attn -> bf16 qkv buffer
  gemm_kernel<0><<<dim3(D3_ / 128, M_ / 128), 256, 0, stream>>>(
      ws + OFF_XB, ws + OFF_WA, ba, ws + OFF_QKV, nullptr, M_, D3_, D_);

  // attention: 2 branches x B x H x NB workgroups
  attn_kernel<<<dim3(NB_, H_, 2 * B_), 128, 0, stream>>>(ws + OFF_QKV, ws + OFF_AO);

  // projection: [5120 x 768] @ [768 x 768] + b_proj -> fp32 d_out (branch0 then branch1)
  gemm_kernel<1><<<dim3(D_ / 128, M_ / 128), 256, 0, stream>>>(
      ws + OFF_AO, ws + OFF_WP, bp, nullptr, out, M_, D_, D_);
}
